// FullDualPathAttentionTFF_82910048682141
// MI455X (gfx1250) — compile-verified, matrix operand order checked by reading, not run on hardware
//
#include <hip/hip_runtime.h>
#include <hip/hip_bf16.h>

typedef __bf16 bf16;
typedef __attribute__((ext_vector_type(16))) __bf16 v16bf;
typedef __attribute__((ext_vector_type(8)))  float  v8f;

#define BB   2
#define NN   2048
#define CC   256
#define HH   8
#define HDIM 32
#define EPSV 1e-5f

// ---------------------------------------------------------------------------
// Fragment loader: per-lane layout for 16-bit WMMA operands (wave32).
// A (16xK, row-major rows) : lane&15 selects row, (lane>>4)*8 selects K-half.
// B (Kx16) loaded from the transposed (column-contiguous) array identically.
// elements [0..7] = K = khalf+0..7 ; [8..15] = K = 16+khalf+0..7
// ---------------------------------------------------------------------------
__device__ __forceinline__ v16bf load_frag16(const bf16* base, int ld) {
  int lane = threadIdx.x & 31;
  const bf16* p = base + (size_t)(lane & 15) * ld + ((lane >> 4) << 3);
  v16bf f;
#pragma unroll
  for (int j = 0; j < 8; ++j) { f[j] = p[j]; f[j + 8] = p[16 + j]; }
  return f;
}

__device__ __forceinline__ v8f wmma_bf16(v16bf a, v16bf b, v8f c) {
  return __builtin_amdgcn_wmma_f32_16x16x32_bf16(false, a, false, b, (short)0, c,
                                                 false, false);
}

__device__ __forceinline__ v8f zero_v8f() {
  v8f z;
#pragma unroll
  for (int i = 0; i < 8; ++i) z[i] = 0.0f;
  return z;
}

// ---------------------------------------------------------------------------
// fp32 -> bf16 weight conversion
// ---------------------------------------------------------------------------
__global__ void cvt_bf16_kernel(const float* __restrict__ src,
                                bf16* __restrict__ dst, int n) {
  int i = blockIdx.x * blockDim.x + threadIdx.x;
  if (i < n) dst[i] = (bf16)src[i];
}

// ---------------------------------------------------------------------------
// Pre-LayerNorm of featA over channels + transpose featA_norm / featB to
// point-major [b][n][c]. One wave (32 lanes) per point, 8 waves per block.
// featA_norm bf16 goes directly into fused[:, :, 0:256].
// ---------------------------------------------------------------------------
__global__ void ln_pre_kernel(const float* __restrict__ featA,
                              const float* __restrict__ featB,
                              const float* __restrict__ gvec,
                              const float* __restrict__ bvec,
                              float* __restrict__ An,     // [b][n][256] f32
                              bf16*  __restrict__ fused,  // [b][n][768]
                              bf16*  __restrict__ Bf) {   // [b][n][256]
  int lane = threadIdx.x & 31, wave = threadIdx.x >> 5;
  int p = blockIdx.x * 8 + wave;
  if (p >= BB * NN) return;
  int b = p / NN, n = p % NN;
  const float* colA = featA + (size_t)b * CC * NN + n;
  float x[8], s = 0.f;
#pragma unroll
  for (int j = 0; j < 8; ++j) {
    x[j] = colA[(size_t)(lane + 32 * j) * NN];
    s += x[j];
  }
#pragma unroll
  for (int m = 1; m < 32; m <<= 1) s += __shfl_xor(s, m, 32);
  float mean = s * (1.0f / CC);
  float vs = 0.f;
#pragma unroll
  for (int j = 0; j < 8; ++j) { float d = x[j] - mean; vs += d * d; }
#pragma unroll
  for (int m = 1; m < 32; m <<= 1) vs += __shfl_xor(vs, m, 32);
  float inv = rsqrtf(vs * (1.0f / CC) + EPSV);

  float* Arow = An + (size_t)p * CC;
  bf16* Frow = fused + (size_t)p * (3 * CC);
  const float* colB = featB + (size_t)b * CC * NN + n;
  bf16* Brow = Bf + (size_t)p * CC;
#pragma unroll
  for (int j = 0; j < 8; ++j) {
    int c = lane + 32 * j;
    float y = (x[j] - mean) * inv * gvec[c] + bvec[c];
    Arow[c] = y;
    Frow[c] = (bf16)y;
    Brow[c] = (bf16)colB[(size_t)c * NN];
  }
}

// ---------------------------------------------------------------------------
// GEMM  O[o,n] = W[o,:] . X[n,:] + bias[o]   (X point-major, ld = ldx)
// store q/k layout: out[((b*H + o/32)*N + n)*32 + o%32]
// ---------------------------------------------------------------------------
__global__ void gemm_qk_kernel(const bf16* __restrict__ W,
                               const bf16* __restrict__ X, int ldx,
                               const float* __restrict__ bias,
                               bf16* __restrict__ out) {
  int lane = threadIdx.x & 31, wave = threadIdx.x >> 5;
  const int tilesPB = (CC / 16) * (NN / 16);
  int t = blockIdx.x * 8 + wave;
  if (t >= BB * tilesPB) return;
  int b = t / tilesPB, r = t % tilesPB;
  int o0 = (r % (CC / 16)) * 16, n0 = (r / (CC / 16)) * 16;
  const bf16* Xb = X + (size_t)b * NN * ldx;
  v8f acc = zero_v8f();
#pragma unroll
  for (int kk = 0; kk < CC; kk += 32) {
    v16bf a = load_frag16(W + (size_t)o0 * CC + kk, CC);
    v16bf bb = load_frag16(Xb + (size_t)n0 * ldx + kk, ldx);
    acc = wmma_bf16(a, bb, acc);
  }
  int col = lane & 15, half = lane >> 4;
#pragma unroll
  for (int g = 0; g < 8; ++g) {
    int o = o0 + g + half * 8, n = n0 + col;
    float v = acc[g] + bias[o];
    out[(((size_t)b * HH + (o >> 5)) * NN + n) * HDIM + (o & 31)] = (bf16)v;
  }
}

// store v layout (GEMM-natural): out[(b*C + o)*N + n]
__global__ void gemm_vT_kernel(const bf16* __restrict__ W,
                               const bf16* __restrict__ X, int ldx,
                               const float* __restrict__ bias,
                               bf16* __restrict__ out) {
  int lane = threadIdx.x & 31, wave = threadIdx.x >> 5;
  const int tilesPB = (CC / 16) * (NN / 16);
  int t = blockIdx.x * 8 + wave;
  if (t >= BB * tilesPB) return;
  int b = t / tilesPB, r = t % tilesPB;
  int o0 = (r % (CC / 16)) * 16, n0 = (r / (CC / 16)) * 16;
  const bf16* Xb = X + (size_t)b * NN * ldx;
  v8f acc = zero_v8f();
#pragma unroll
  for (int kk = 0; kk < CC; kk += 32) {
    v16bf a = load_frag16(W + (size_t)o0 * CC + kk, CC);
    v16bf bb = load_frag16(Xb + (size_t)n0 * ldx + kk, ldx);
    acc = wmma_bf16(a, bb, acc);
  }
  int col = lane & 15, half = lane >> 4;
#pragma unroll
  for (int g = 0; g < 8; ++g) {
    int o = o0 + g + half * 8, n = n0 + col;
    out[((size_t)b * CC + o) * NN + n] = (bf16)(acc[g] + bias[o]);
  }
}

// row-major point output: out[(b*N+n)*ldo + o]; epilogue = bias OR BN+ReLU
__global__ void gemm_row_kernel(const bf16* __restrict__ W,
                                const bf16* __restrict__ X, int ldx,
                                int Cout, int Cin,
                                const float* __restrict__ bias,
                                bf16* __restrict__ out, int ldo, int bnrelu,
                                const float* __restrict__ bg,
                                const float* __restrict__ bb,
                                const float* __restrict__ bm,
                                const float* __restrict__ bv) {
  int lane = threadIdx.x & 31, wave = threadIdx.x >> 5;
  const int tilesPB = (Cout / 16) * (NN / 16);
  int t = blockIdx.x * 8 + wave;
  if (t >= BB * tilesPB) return;
  int b = t / tilesPB, r = t % tilesPB;
  int o0 = (r % (Cout / 16)) * 16, n0 = (r / (Cout / 16)) * 16;
  const bf16* Xb = X + (size_t)b * NN * ldx;
  v8f acc = zero_v8f();
  for (int kk = 0; kk < Cin; kk += 32) {
    v16bf a = load_frag16(W + (size_t)o0 * Cin + kk, Cin);
    v16bf bbf = load_frag16(Xb + (size_t)n0 * ldx + kk, ldx);
    acc = wmma_bf16(a, bbf, acc);
  }
  int col = lane & 15, half = lane >> 4;
#pragma unroll
  for (int g = 0; g < 8; ++g) {
    int o = o0 + g + half * 8, n = n0 + col;
    float v = acc[g];
    if (bnrelu) {
      float inv = rsqrtf(bv[o] + EPSV);
      v = (v - bm[o]) * inv * bg[o] + bb[o];
      v = fmaxf(v, 0.0f);
    } else {
      v += bias[o];
    }
    out[(size_t)(b * NN + n) * ldo + o] = (bf16)v;
  }
}

// per-point sum of squares of qd / kd rows (for cdist expansion)
__global__ void sumsq_kernel(const bf16* __restrict__ qd,
                             const bf16* __restrict__ kd,
                             float* __restrict__ sqq, float* __restrict__ sqk) {
  int lane = threadIdx.x & 31, wave = threadIdx.x >> 5;
  int p = blockIdx.x * 8 + wave;
  if (p >= BB * NN) return;
  const bf16* r0 = qd + (size_t)p * CC;
  const bf16* r1 = kd + (size_t)p * CC;
  float s0 = 0.f, s1 = 0.f;
#pragma unroll
  for (int j = 0; j < 8; ++j) {
    float a = (float)r0[lane + 32 * j];
    float c = (float)r1[lane + 32 * j];
    s0 += a * a; s1 += c * c;
  }
#pragma unroll
  for (int m = 1; m < 32; m <<= 1) {
    s0 += __shfl_xor(s0, m, 32);
    s1 += __shfl_xor(s1, m, 32);
  }
  if (lane == 0) { sqq[p] = s0; sqk[p] = s1; }
}

// ---------------------------------------------------------------------------
// Feature cdist matrix: D[b,n,m] = sqrt(max(|qd|^2+|kd|^2-2 qd.kd, 0))
// WMMA Gram kernel over K = C = 256.
// ---------------------------------------------------------------------------
__global__ void dissim_kernel(const bf16* __restrict__ qd,
                              const bf16* __restrict__ kd,
                              const float* __restrict__ sqq,
                              const float* __restrict__ sqk,
                              float* __restrict__ D) {
  int lane = threadIdx.x & 31, wave = threadIdx.x >> 5;
  const int tilesPB = (NN / 16) * (NN / 16);
  int t = blockIdx.x * 8 + wave;
  if (t >= BB * tilesPB) return;
  int b = t / tilesPB, r = t % tilesPB;
  int m0 = (r % (NN / 16)) * 16, n0 = (r / (NN / 16)) * 16;
  v8f acc = zero_v8f();
#pragma unroll
  for (int kk = 0; kk < CC; kk += 32) {
    v16bf a = load_frag16(qd + ((size_t)b * NN + n0) * CC + kk, CC);
    v16bf bb = load_frag16(kd + ((size_t)b * NN + m0) * CC + kk, CC);
    acc = wmma_bf16(a, bb, acc);
  }
  int col = lane & 15, half = lane >> 4;
#pragma unroll
  for (int g = 0; g < 8; ++g) {
    int n = n0 + g + half * 8, m = m0 + col;
    float d2 = sqq[b * NN + n] + sqk[b * NN + m] - 2.0f * acc[g];
    D[((size_t)b * NN + n) * NN + m] = sqrtf(fmaxf(d2, 0.0f));
  }
}

// ---------------------------------------------------------------------------
// Fused dual-path flash attention. 1 workgroup = 16 query rows, 8 waves,
// wave h = head h, both softmax paths online; 6 WMMAs per 32-col step.
// ---------------------------------------------------------------------------
__device__ __forceinline__ void flash_update(float* s0, float* s1, float* mrun,
                                             float* lrun, v8f& c0, v8f& c1,
                                             bf16* pl, const bf16* vbase0,
                                             const bf16* vbase1) {
  int lane = threadIdx.x & 31, col = lane & 15, half = lane >> 4;
  float tm[8];
#pragma unroll
  for (int g = 0; g < 8; ++g) tm[g] = fmaxf(s0[g], s1[g]);
#pragma unroll
  for (int m = 1; m < 16; m <<= 1)
#pragma unroll
    for (int g = 0; g < 8; ++g) tm[g] = fmaxf(tm[g], __shfl_xor(tm[g], m, 32));
  float p0[8], p1[8], rs[8], sc[8];
#pragma unroll
  for (int g = 0; g < 8; ++g) {
    float nm = fmaxf(mrun[g], tm[g]);
    sc[g] = __expf(mrun[g] - nm);
    p0[g] = __expf(s0[g] - nm);
    p1[g] = __expf(s1[g] - nm);
    rs[g] = p0[g] + p1[g];
    mrun[g] = nm;
  }
#pragma unroll
  for (int m = 1; m < 16; m <<= 1)
#pragma unroll
    for (int g = 0; g < 8; ++g) rs[g] += __shfl_xor(rs[g], m, 32);
#pragma unroll
  for (int g = 0; g < 8; ++g) {
    lrun[g] = lrun[g] * sc[g] + rs[g];
    c0[g] *= sc[g];
    c1[g] *= sc[g];
    int row = g + half * 8;
    pl[row * 32 + col] = (bf16)p0[g];
    pl[row * 32 + 16 + col] = (bf16)p1[g];
  }
  v16bf pf = load_frag16(pl, 32);
  v16bf vf0 = load_frag16(vbase0, NN);
  v16bf vf1 = load_frag16(vbase1, NN);
  c0 = wmma_bf16(pf, vf0, c0);
  c1 = wmma_bf16(pf, vf1, c1);
}

__global__ void attn_kernel(const bf16* __restrict__ q,
                            const bf16* __restrict__ k,
                            const bf16* __restrict__ vT,
                            const float* __restrict__ xyzA,
                            const float* __restrict__ xyzB,
                            const float* __restrict__ Ddis,
                            const float* __restrict__ sbw1,
                            const float* __restrict__ sbg,
                            const float* __restrict__ sbb,
                            const float* __restrict__ sbm,
                            const float* __restrict__ sbv,
                            const float* __restrict__ sbw2,
                            const float* __restrict__ sbb2,
                            float* __restrict__ ctx_sim,
                            float* __restrict__ ctx_dis) {
  __shared__ bf16 plds[8][16 * 32];
  int lane = threadIdx.x & 31, h = threadIdx.x >> 5;
  int b = blockIdx.x / (NN / 16);
  int n0 = (blockIdx.x % (NN / 16)) * 16;
  int col = lane & 15, half = lane >> 4;

  // fold BN into per-channel affine of the spatial-bias MLP
  float aj[4], cj[4], w2h[4];
#pragma unroll
  for (int j = 0; j < 4; ++j) {
    float inv = rsqrtf(sbv[j] + EPSV);
    aj[j] = -sbw1[j] * sbg[j] * inv;
    cj[j] = sbb[j] - sbm[j] * sbg[j] * inv;
    w2h[j] = sbw2[h * 4 + j];
  }
  float b2h = sbb2[h];

  v16bf qf = load_frag16(q + (((size_t)b * HH + h) * NN + n0) * HDIM, HDIM);

  float ax[8], ay[8], az[8];
  int nr[8];
#pragma unroll
  for (int g = 0; g < 8; ++g) {
    nr[g] = n0 + g + half * 8;
    const float* pa = xyzA + ((size_t)b * NN + nr[g]) * 3;
    ax[g] = pa[0]; ay[g] = pa[1]; az[g] = pa[2];
  }

  float ms[8], ls[8], md[8], ldn[8];
  v8f cs0 = zero_v8f(), cs1 = zero_v8f(), cd0 = zero_v8f(), cd1 = zero_v8f();
#pragma unroll
  for (int g = 0; g < 8; ++g) { ms[g] = -1e30f; md[g] = -1e30f; ls[g] = 0.f; ldn[g] = 0.f; }

  const float scale = 0.17677669529663687f;  // 1/sqrt(HD)
  const bf16* kb = k + ((size_t)b * HH + h) * NN * HDIM;
  const bf16* vb = vT + (size_t)(b * CC + h * HDIM) * NN;

  for (int m0 = 0; m0 < NN; m0 += 32) {
    v16bf kf0 = load_frag16(kb + (size_t)m0 * HDIM, HDIM);
    v16bf kf1 = load_frag16(kb + (size_t)(m0 + 16) * HDIM, HDIM);
    v8f sv0 = wmma_bf16(qf, kf0, zero_v8f());
    v8f sv1 = wmma_bf16(qf, kf1, zero_v8f());

    int mA = m0 + col, mB = m0 + 16 + col;
    const float* pb0 = xyzB + ((size_t)b * NN + mA) * 3;
    const float* pb1 = xyzB + ((size_t)b * NN + mB) * 3;
    float bx0 = pb0[0], by0 = pb0[1], bz0 = pb0[2];
    float bx1 = pb1[0], by1 = pb1[1], bz1 = pb1[2];

    float s0[8], s1[8], d0[8], d1[8];
#pragma unroll
    for (int g = 0; g < 8; ++g) {
      size_t drow = ((size_t)b * NN + nr[g]) * NN;
      float dx = ax[g] - bx0, dy = ay[g] - by0, dz = az[g] - bz0;
      float dist0 = sqrtf(dx * dx + dy * dy + dz * dz);
      dx = ax[g] - bx1; dy = ay[g] - by1; dz = az[g] - bz1;
      float dist1 = sqrtf(dx * dx + dy * dy + dz * dz);
      float bias0 = b2h, bias1 = b2h;
#pragma unroll
      for (int j = 0; j < 4; ++j) {
        bias0 += w2h[j] * fmaxf(aj[j] * dist0 + cj[j], 0.0f);
        bias1 += w2h[j] * fmaxf(aj[j] * dist1 + cj[j], 0.0f);
      }
      s0[g] = sv0[g] * scale + bias0;
      s1[g] = sv1[g] * scale + bias1;
      d0[g] = Ddis[drow + mA] + bias0;
      d1[g] = Ddis[drow + mB] + bias1;
    }
    flash_update(s0, s1, ms, ls, cs0, cs1, &plds[h][0],
                 vb + m0, vb + (size_t)16 * NN + m0);
    flash_update(d0, d1, md, ldn, cd0, cd1, &plds[h][0],
                 vb + m0, vb + (size_t)16 * NN + m0);
  }

#pragma unroll
  for (int g = 0; g < 8; ++g) {
    float is = 1.0f / ls[g], idn = 1.0f / ldn[g];
    size_t base = ((size_t)b * NN + nr[g]) * CC + h * HDIM;
    ctx_sim[base + col] = cs0[g] * is;
    ctx_sim[base + 16 + col] = cs1[g] * is;
    ctx_dis[base + col] = cd0[g] * idn;
    ctx_dis[base + 16 + col] = cd1[g] * idn;
  }
}

// ---------------------------------------------------------------------------
// Post-LayerNorm of a [b][n][256] f32 tensor, packing bf16 into fused[...,off]
// ---------------------------------------------------------------------------
__global__ void postln_kernel(const float* __restrict__ ctx,
                              const float* __restrict__ gvec,
                              const float* __restrict__ bvec,
                              bf16* __restrict__ fused, int off) {
  int lane = threadIdx.x & 31, wave = threadIdx.x >> 5;
  int p = blockIdx.x * 8 + wave;
  if (p >= BB * NN) return;
  const float* row = ctx + (size_t)p * CC;
  float x[8], s = 0.f;
#pragma unroll
  for (int j = 0; j < 8; ++j) { x[j] = row[lane + 32 * j]; s += x[j]; }
#pragma unroll
  for (int m = 1; m < 32; m <<= 1) s += __shfl_xor(s, m, 32);
  float mean = s * (1.0f / CC);
  float vs = 0.f;
#pragma unroll
  for (int j = 0; j < 8; ++j) { float d = x[j] - mean; vs += d * d; }
#pragma unroll
  for (int m = 1; m < 32; m <<= 1) vs += __shfl_xor(vs, m, 32);
  float inv = rsqrtf(vs * (1.0f / CC) + EPSV);
  bf16* Frow = fused + (size_t)p * (3 * CC) + off;
#pragma unroll
  for (int j = 0; j < 8; ++j) {
    int c = lane + 32 * j;
    Frow[c] = (bf16)((x[j] - mean) * inv * gvec[c] + bvec[c]);
  }
}

// ---------------------------------------------------------------------------
// Final GEMM: out[b,o,n] = fw2 . hid + fb2 + featA (residual)
// ---------------------------------------------------------------------------
__global__ void gemm_final_kernel(const bf16* __restrict__ W,
                                  const bf16* __restrict__ X,  // hid rows, ld 512
                                  const float* __restrict__ bias,
                                  const float* __restrict__ featA,
                                  float* __restrict__ out) {
  int lane = threadIdx.x & 31, wave = threadIdx.x >> 5;
  const int tilesPB = (CC / 16) * (NN / 16);
  int t = blockIdx.x * 8 + wave;
  if (t >= BB * tilesPB) return;
  int b = t / tilesPB, r = t % tilesPB;
  int o0 = (r % (CC / 16)) * 16, n0 = (r / (CC / 16)) * 16;
  const bf16* Xb = X + (size_t)b * NN * (2 * CC);
  v8f acc = zero_v8f();
#pragma unroll
  for (int kk = 0; kk < 2 * CC; kk += 32) {
    v16bf a = load_frag16(W + (size_t)o0 * (2 * CC) + kk, 2 * CC);
    v16bf bb = load_frag16(Xb + (size_t)n0 * (2 * CC) + kk, 2 * CC);
    acc = wmma_bf16(a, bb, acc);
  }
  int col = lane & 15, half = lane >> 4;
#pragma unroll
  for (int g = 0; g < 8; ++g) {
    int o = o0 + g + half * 8, n = n0 + col;
    size_t idx = ((size_t)b * CC + o) * NN + n;
    out[idx] = acc[g] + bias[o] + featA[idx];
  }
}

// ---------------------------------------------------------------------------
extern "C" void kernel_launch(void* const* d_in, const int* in_sizes, int n_in,
                              void* d_out, int out_size, void* d_ws,
                              size_t ws_size, hipStream_t stream) {
  const float* xyz_A = (const float*)d_in[0];
  const float* featA = (const float*)d_in[1];
  const float* xyz_B = (const float*)d_in[2];
  const float* featB = (const float*)d_in[3];
  const float* qw = (const float*)d_in[4];  const float* qb = (const float*)d_in[5];
  const float* kw = (const float*)d_in[6];  const float* kbv = (const float*)d_in[7];
  const float* vw = (const float*)d_in[8];  const float* vbv = (const float*)d_in[9];
  const float* sbw1 = (const float*)d_in[10];
  const float* sbg = (const float*)d_in[11];
  const float* sbb = (const float*)d_in[12];
  const float* sbm = (const float*)d_in[13];
  const float* sbv = (const float*)d_in[14];
  const float* sbw2 = (const float*)d_in[15];
  const float* sbb2 = (const float*)d_in[16];
  const float* qdw = (const float*)d_in[17]; const float* qdb = (const float*)d_in[18];
  const float* kdw = (const float*)d_in[19]; const float* kdb = (const float*)d_in[20];
  const float* lin_g = (const float*)d_in[21]; const float* lin_b = (const float*)d_in[22];
  const float* lsim_g = (const float*)d_in[23]; const float* lsim_b = (const float*)d_in[24];
  const float* ldis_g = (const float*)d_in[25]; const float* ldis_b = (const float*)d_in[26];
  const float* fw1 = (const float*)d_in[27];
  const float* fbg = (const float*)d_in[28]; const float* fbb = (const float*)d_in[29];
  const float* fbm = (const float*)d_in[30]; const float* fbv = (const float*)d_in[31];
  const float* fw2 = (const float*)d_in[32]; const float* fb2 = (const float*)d_in[33];

  // workspace carve-up (256B aligned)
  char* ws = (char*)d_ws;
  size_t off = 0;
  auto alloc = [&](size_t bytes) {
    void* p = ws + off;
    off += (bytes + 255) & ~(size_t)255;
    return p;
  };
  const size_t P = (size_t)BB * NN;
  float* An      = (float*)alloc(P * CC * 4);
  bf16*  fused   = (bf16*) alloc(P * 3 * CC * 2);
  bf16*  Bf      = (bf16*) alloc(P * CC * 2);
  bf16*  qbuf    = (bf16*) alloc(P * CC * 2);
  bf16*  kbuf    = (bf16*) alloc(P * CC * 2);
  bf16*  vbuf    = (bf16*) alloc(P * CC * 2);
  bf16*  qdbuf   = (bf16*) alloc(P * CC * 2);
  bf16*  kdbuf   = (bf16*) alloc(P * CC * 2);
  float* sqq     = (float*)alloc(P * 4);
  float* sqk     = (float*)alloc(P * 4);
  float* Ddis    = (float*)alloc((size_t)BB * NN * NN * 4);
  float* ctxs    = (float*)alloc(P * CC * 4);
  float* ctxd    = (float*)alloc(P * CC * 4);
  bf16*  hid     = (bf16*) alloc(P * 2 * CC * 2);
  bf16*  wq  = (bf16*)alloc(CC * CC * 2);
  bf16*  wk  = (bf16*)alloc(CC * CC * 2);
  bf16*  wv  = (bf16*)alloc(CC * CC * 2);
  bf16*  wqd = (bf16*)alloc(CC * CC * 2);
  bf16*  wkd = (bf16*)alloc(CC * CC * 2);
  bf16*  wf1 = (bf16*)alloc((size_t)2 * CC * 3 * CC * 2);
  bf16*  wf2 = (bf16*)alloc((size_t)CC * 2 * CC * 2);

  // 1) weights -> bf16
  auto cvt = [&](const float* s, bf16* d, int n) {
    cvt_bf16_kernel<<<(n + 255) / 256, 256, 0, stream>>>(s, d, n);
  };
  cvt(qw, wq, CC * CC);  cvt(kw, wk, CC * CC);  cvt(vw, wv, CC * CC);
  cvt(qdw, wqd, CC * CC); cvt(kdw, wkd, CC * CC);
  cvt(fw1, wf1, 2 * CC * 3 * CC); cvt(fw2, wf2, CC * 2 * CC);

  // 2) pre-LN + transposes
  ln_pre_kernel<<<(int)(P / 8), 256, 0, stream>>>(featA, featB, lin_g, lin_b,
                                                  An, fused, Bf);

  // 3) projections
  const int gemmCC_blocks = BB * (CC / 16) * (NN / 16) / 8;  // 512
  gemm_qk_kernel<<<gemmCC_blocks, 256, 0, stream>>>(wq, fused, 3 * CC, qb, qbuf);
  gemm_qk_kernel<<<gemmCC_blocks, 256, 0, stream>>>(wk, Bf, CC, kbv, kbuf);
  gemm_vT_kernel<<<gemmCC_blocks, 256, 0, stream>>>(wv, Bf, CC, vbv, vbuf);
  gemm_row_kernel<<<gemmCC_blocks, 256, 0, stream>>>(wqd, fused, 3 * CC, CC, CC,
                                                     qdb, qdbuf, CC, 0,
                                                     nullptr, nullptr, nullptr, nullptr);
  gemm_row_kernel<<<gemmCC_blocks, 256, 0, stream>>>(wkd, Bf, CC, CC, CC,
                                                     kdb, kdbuf, CC, 0,
                                                     nullptr, nullptr, nullptr, nullptr);

  // 4) row sumsq + feature cdist matrix
  sumsq_kernel<<<(int)(P / 8), 256, 0, stream>>>(qdbuf, kdbuf, sqq, sqk);
  dissim_kernel<<<BB * (NN / 16) * (NN / 16) / 8, 256, 0, stream>>>(
      qdbuf, kdbuf, sqq, sqk, Ddis);

  // 5) fused dual-path flash attention
  attn_kernel<<<BB * (NN / 16), 256, 0, stream>>>(
      qbuf, kbuf, vbuf, xyz_A, xyz_B, Ddis, sbw1, sbg, sbb, sbm, sbv, sbw2,
      sbb2, ctxs, ctxd);

  // 6) post-LN -> fused[:,256:512] and [:,512:768]
  postln_kernel<<<(int)(P / 8), 256, 0, stream>>>(ctxs, lsim_g, lsim_b, fused, CC);
  postln_kernel<<<(int)(P / 8), 256, 0, stream>>>(ctxd, ldis_g, ldis_b, fused, 2 * CC);

  // 7) fusion MLP: 3C->2C (BN+ReLU) then 2C->C (+bias +residual)
  gemm_row_kernel<<<BB * (2 * CC / 16) * (NN / 16) / 8, 256, 0, stream>>>(
      wf1, fused, 3 * CC, 2 * CC, 3 * CC, nullptr, hid, 2 * CC, 1,
      fbg, fbb, fbm, fbv);
  gemm_final_kernel<<<gemmCC_blocks, 256, 0, stream>>>(wf2, hid, fb2, featA,
                                                       (float*)d_out);
}
